// TransformerBlock_76974403879613
// MI455X (gfx1250) — matrix operand order repaired and checked by reading, not compiled, not run on hardware
//
#include <hip/hip_runtime.h>
#include <math.h>

// ---------------------------------------------------------------------------
// Transformer block on MI455X (gfx1250, wave32, WMMA).
// GEMMs + attention via v_wmma_f32_16x16x32_bf16 (bf16 in, f32 acc).
// GEMM A-tiles staged with global_load_async_to_lds_b128 (ASYNCcnt path),
// double-buffered so the DMA overlaps the WMMA work.
// ---------------------------------------------------------------------------

typedef __bf16 bf16;
typedef __attribute__((ext_vector_type(16))) __bf16 v16bf;
typedef __attribute__((ext_vector_type(8)))  __bf16 v8bf;
typedef __attribute__((ext_vector_type(4)))  __bf16 v4bf;
typedef __attribute__((ext_vector_type(8)))  float  v8f;
typedef __attribute__((ext_vector_type(4)))  float  v4f;

#define DMODEL 1024
#define NHEADS 16
#define DHEAD  64
#define DFF    4096
#define SEQ    2048
#define BATCH  2
#define NTOK   (BATCH * SEQ)   // 4096

__device__ __forceinline__ bf16 f2bf(float f) {
    union { float f; unsigned u; } in;
    in.f = f;
    unsigned u = in.u;
    u += 0x7FFFu + ((u >> 16) & 1u);      // round-to-nearest-even
    union { unsigned short s; bf16 b; } out;
    out.s = (unsigned short)(u >> 16);
    return out.b;
}

__device__ __forceinline__ v8f vzero8() {
    v8f z;
#pragma unroll
    for (int i = 0; i < 8; ++i) z[i] = 0.0f;
    return z;
}

__device__ __forceinline__ v16bf cat16(v8bf lo, v8bf hi) {
    return __builtin_shufflevector(lo, hi, 0, 1, 2, 3, 4, 5, 6, 7,
                                           8, 9, 10, 11, 12, 13, 14, 15);
}

__device__ __forceinline__ v8f wmma_bf16(v16bf a, v16bf b, v8f c) {
    // D = A(16x32 bf16) * B(32x16 bf16) + C(16x16 f32)
    return __builtin_amdgcn_wmma_f32_16x16x32_bf16(
        false, a, false, b, (short)0, c, false, false);
}

// Async memory->LDS copy of 16 bytes per lane (tracked with ASYNCcnt).
__device__ __forceinline__ void async_copy_b128(unsigned lds_byte_off,
                                                const void* gaddr) {
    asm volatile("global_load_async_to_lds_b128 %0, %1, off"
                 :: "v"(lds_byte_off),
                    "v"((unsigned long long)(uintptr_t)gaddr)
                 : "memory");
}
__device__ __forceinline__ void wait_asynccnt0() {
    asm volatile("s_wait_asynccnt 0" ::: "memory");
}

// ---------------------------------------------------------------------------
// fp32 -> bf16 conversion
// ---------------------------------------------------------------------------
__global__ __launch_bounds__(256) void cvt_f32_bf16(const float* __restrict__ in,
                                                    bf16* __restrict__ out, int n4) {
    int i = blockIdx.x * 256 + threadIdx.x;
    if (i < n4) {
        v4f v = *(const v4f*)(in + (size_t)i * 4);
        v4bf o;
#pragma unroll
        for (int e = 0; e < 4; ++e) o[e] = f2bf(v[e]);
        *(v4bf*)(out + (size_t)i * 4) = o;
    }
}

// ---------------------------------------------------------------------------
// RMSNorm: one 256-thread block per token row (1024 cols), bf16 output
// ---------------------------------------------------------------------------
__global__ __launch_bounds__(256) void rmsnorm_bf16(const float* __restrict__ x,
                                                    const float* __restrict__ g,
                                                    bf16* __restrict__ out) {
    __shared__ float red[8];
    const int row = blockIdx.x;
    const int tid = threadIdx.x;
    const int lane = tid & 31;
    const int w = tid >> 5;

    const float* xr = x + (size_t)row * DMODEL;
    v4f v = *(const v4f*)(xr + tid * 4);
    float ss = v[0] * v[0] + v[1] * v[1] + v[2] * v[2] + v[3] * v[3];
#pragma unroll
    for (int m = 1; m < 32; m <<= 1) ss += __shfl_xor(ss, m, 32);
    if (lane == 0) red[w] = ss;
    __syncthreads();
    float tot = 0.0f;
#pragma unroll
    for (int i = 0; i < 8; ++i) tot += red[i];
    const float inv = rsqrtf(tot * (1.0f / (float)DMODEL) + 1e-5f);

    v4f gg = *(const v4f*)(g + tid * 4);
    v4bf o;
#pragma unroll
    for (int e = 0; e < 4; ++e) o[e] = f2bf(v[e] * gg[e] * inv);
    *(v4bf*)(out + (size_t)row * DMODEL + tid * 4) = o;
}

// ---------------------------------------------------------------------------
// Transpose V: [b*2048+s][h*64+d] -> [((b*16+h)*64+d)*2048 + s]
// ---------------------------------------------------------------------------
__global__ __launch_bounds__(256) void transpose_v(const bf16* __restrict__ vb,
                                                   bf16* __restrict__ vt) {
    size_t i = (size_t)blockIdx.x * 256 + threadIdx.x;   // 4M elements
    int s = (int)(i & 2047);
    size_t r = i >> 11;
    int d = (int)(r & 63);  r >>= 6;
    int h = (int)(r & 15);
    int b = (int)(r >> 4);
    vt[i] = vb[((size_t)(b * SEQ + s)) * DMODEL + h * DHEAD + d];
}

// ---------------------------------------------------------------------------
// WMMA GEMM: C[M,N] = A[M,K] @ W[N,K]^T   (A,W bf16 row-major, fp32 accum)
// Block 256 thr = 8 waves, block tile 128x128, wave tile 64x32 (4x2 subtiles).
// A double-buffered in LDS via async global->LDS DMA; B fragments contiguous
// from global (16 contiguous K per lane, same k order as the A fragments).
// EPI: 0 = store bf16, 1 = GELU + store bf16, 2 = fp32 residual add
// ---------------------------------------------------------------------------
template <int EPI>
__global__ __launch_bounds__(256) void gemm_wmma_bf16(
    const bf16* __restrict__ A, const bf16* __restrict__ W,
    bf16* __restrict__ obf, float* __restrict__ of32,
    const float* __restrict__ resid, int M, int N, int K) {
    __shared__ bf16 As[2][128 * 40];   // padded stride 40 bf16 (80 B)

    const int tid  = threadIdx.x;
    const int lane = tid & 31;
    const int w    = tid >> 5;
    const int wm   = w >> 2;         // 0..1
    const int wn   = w & 3;          // 0..3
    const int lo16 = lane & 15;
    const int hi   = lane >> 4;

    const int bm = blockIdx.y * 128;
    const int bn = blockIdx.x * 128;

    v8f acc[4][2];
#pragma unroll
    for (int i = 0; i < 4; ++i)
#pragma unroll
        for (int j = 0; j < 2; ++j) acc[i][j] = vzero8();

    const bf16* brow[2];
#pragma unroll
    for (int j = 0; j < 2; ++j)
        brow[j] = W + (size_t)(bn + wn * 32 + j * 16 + lo16) * K;

    // A staging: each thread DMAs two 16-byte chunks of the 128x32 tile.
    const int r0 = tid >> 2;         // rows 0..63
    const int r1 = r0 + 64;          // rows 64..127
    const int cc = (tid & 3) * 8;    // col chunk

    const bf16* arow0 = A + (size_t)(bm + r0) * K + cc;
    const bf16* arow1 = A + (size_t)(bm + r1) * K + cc;
    const unsigned lbase0 = (unsigned)(uintptr_t)&As[0][r0 * 40 + cc];
    const unsigned lbase1 = (unsigned)(uintptr_t)&As[0][r1 * 40 + cc];
    const unsigned bufstride = (unsigned)(128 * 40 * sizeof(bf16));

    // prologue: stage k0 = 0 into buffer 0
    async_copy_b128(lbase0, arow0);
    async_copy_b128(lbase1, arow1);

    int buf = 0;
    for (int k0 = 0; k0 < K; k0 += 32) {
        wait_asynccnt0();            // own DMA for As[buf] complete
        __syncthreads();             // everyone's DMA done; prev buf reads done

        if (k0 + 32 < K) {           // stage next tile into the other buffer
            const unsigned l = bufstride * (unsigned)(buf ^ 1);
            async_copy_b128(lbase0 + l, arow0 + k0 + 32);
            async_copy_b128(lbase1 + l, arow1 + k0 + 32);
        }

        // streaming B prefetch ~8 iterations ahead (global_prefetch_b8)
        __builtin_prefetch(brow[0] + k0 + 256, 0, 0);
        __builtin_prefetch(brow[1] + k0 + 256, 0, 0);

        v16bf bfrag[2];
#pragma unroll
        for (int j = 0; j < 2; ++j) {
            const bf16* p = brow[j] + k0 + hi * 16;
            bfrag[j] = cat16(*(const v8bf*)p, *(const v8bf*)(p + 8));
        }
#pragma unroll
        for (int i = 0; i < 4; ++i) {
            const bf16* ap = &As[buf][(wm * 64 + i * 16 + lo16) * 40];
            v16bf af = cat16(*(const v8bf*)(ap + hi * 16),
                             *(const v8bf*)(ap + hi * 16 + 8));
#pragma unroll
            for (int j = 0; j < 2; ++j)
                acc[i][j] = wmma_bf16(af, bfrag[j], acc[i][j]);
        }
        buf ^= 1;
    }

#pragma unroll
    for (int i = 0; i < 4; ++i)
#pragma unroll
        for (int j = 0; j < 2; ++j)
#pragma unroll
            for (int r = 0; r < 8; ++r) {
                const int row = bm + wm * 64 + i * 16 + r + 8 * hi;
                const int col = bn + wn * 32 + j * 16 + lo16;
                const size_t idx = (size_t)row * N + col;
                float v = acc[i][j][r];
                if (EPI == 0) {
                    obf[idx] = f2bf(v);
                } else if (EPI == 1) {
                    float gv = 0.5f * v * (1.0f + erff(v * 0.70710678118f));
                    obf[idx] = f2bf(gv);
                } else {
                    of32[idx] = resid[idx] + v;
                }
            }
}

// ---------------------------------------------------------------------------
// Flash attention: one wave per 16-query tile, 32-key chunks, causal.
// QK^T and PV both via v_wmma_f32_16x16x32_bf16. V is pre-transposed.
// ---------------------------------------------------------------------------
__global__ __launch_bounds__(256) void attn_wmma(const bf16* __restrict__ qb,
                                                 const bf16* __restrict__ kb,
                                                 const bf16* __restrict__ vt,
                                                 bf16* __restrict__ ob) {
    __shared__ bf16 P[8][16 * 40];   // per-wave 16x32 P tile, padded stride

    const int tid  = threadIdx.x;
    const int lane = tid & 31;
    const int w    = tid >> 5;
    const int lo16 = lane & 15;
    const int hi   = lane >> 4;

    const int t  = blockIdx.x * 8 + w;         // 4096 tiles
    const int qt = t & 127;
    const int hh = (t >> 7) & 15;
    const int b  = t >> 11;
    const int q0 = qt * 16;

    // Q fragments (A operand): rows q0..q0+15, K-dim = d (two chunks of 32)
    const bf16* qrow = qb + (size_t)(b * SEQ + q0 + lo16) * DMODEL + hh * DHEAD;
    v16bf qf[2];
#pragma unroll
    for (int c = 0; c < 2; ++c)
        qf[c] = cat16(*(const v8bf*)(qrow + c * 32 + hi * 16),
                      *(const v8bf*)(qrow + c * 32 + hi * 16 + 8));

    float run_m[8], run_l[8];
#pragma unroll
    for (int r = 0; r < 8; ++r) { run_m[r] = -1e30f; run_l[r] = 0.0f; }
    v8f o[4];
#pragma unroll
    for (int i = 0; i < 4; ++i) o[i] = vzero8();

    const int nch = (qt >> 1) + 1;             // causal: keys <= q only
    for (int ch = 0; ch < nch; ++ch) {
        const int kb0 = ch * 32;
        v8f s[2] = { vzero8(), vzero8() };
#pragma unroll
        for (int dc = 0; dc < 2; ++dc) {
#pragma unroll
            for (int j = 0; j < 2; ++j) {
                const bf16* kp = kb + (size_t)(b * SEQ + kb0 + j * 16 + lo16) * DMODEL
                                    + hh * DHEAD + dc * 32 + hi * 16;
                v16bf bf = cat16(*(const v8bf*)kp, *(const v8bf*)(kp + 8));
                s[j] = wmma_bf16(qf[dc], bf, s[j]);
            }
        }

        const bool last = (ch == nch - 1);
#pragma unroll
        for (int r = 0; r < 8; ++r) {
            const int q = q0 + r + 8 * hi;
            float v0 = s[0][r] * 0.125f;                  // 1/sqrt(64)
            float v1 = s[1][r] * 0.125f;
            if (last) {
                if (kb0 + lo16 > q)      v0 = -1e30f;
                if (kb0 + 16 + lo16 > q) v1 = -1e30f;
            }
            float mt = fmaxf(v0, v1);
#pragma unroll
            for (int m = 1; m < 16; m <<= 1) mt = fmaxf(mt, __shfl_xor(mt, m, 32));
            const float nm = fmaxf(run_m[r], mt);
            const float f  = __expf(run_m[r] - nm);
            const float p0 = __expf(v0 - nm);
            const float p1 = __expf(v1 - nm);
            float rs = p0 + p1;
#pragma unroll
            for (int m = 1; m < 16; m <<= 1) rs += __shfl_xor(rs, m, 32);
            run_l[r] = run_l[r] * f + rs;
            run_m[r] = nm;
#pragma unroll
            for (int tt = 0; tt < 4; ++tt) o[tt][r] *= f;
            bf16* pr = &P[w][(r + 8 * hi) * 40];
            pr[lo16]      = f2bf(p0);
            pr[16 + lo16] = f2bf(p1);
        }

        // P as A operand (16x32), V chunk as B operand (32x16 per d-subtile)
        const bf16* pm = &P[w][lo16 * 40];
        v16bf pf = cat16(*(const v8bf*)(pm + hi * 16),
                         *(const v8bf*)(pm + hi * 16 + 8));
#pragma unroll
        for (int tt = 0; tt < 4; ++tt) {
            const bf16* vp = vt + ((size_t)(b * NHEADS + hh) * DHEAD + tt * 16 + lo16) * SEQ
                                + kb0 + hi * 16;
            v16bf bf = cat16(*(const v8bf*)vp, *(const v8bf*)(vp + 8));
            o[tt] = wmma_bf16(pf, bf, o[tt]);
        }
    }

#pragma unroll
    for (int r = 0; r < 8; ++r) {
        const float inv = 1.0f / run_l[r];
        const int tok = b * SEQ + q0 + r + 8 * hi;
#pragma unroll
        for (int tt = 0; tt < 4; ++tt)
            ob[(size_t)tok * DMODEL + hh * DHEAD + tt * 16 + lo16] =
                f2bf(o[tt][r] * inv);
    }
}

// ---------------------------------------------------------------------------
// Host orchestration
// ---------------------------------------------------------------------------
extern "C" void kernel_launch(void* const* d_in, const int* in_sizes, int n_in,
                              void* d_out, int out_size, void* d_ws, size_t ws_size,
                              hipStream_t stream) {
    const float* x  = (const float*)d_in[0];
    const float* Wq = (const float*)d_in[1];
    const float* Wk = (const float*)d_in[2];
    const float* Wv = (const float*)d_in[3];
    const float* Wo = (const float*)d_in[4];
    const float* W1 = (const float*)d_in[5];
    const float* W2 = (const float*)d_in[6];
    const float* g1 = (const float*)d_in[7];
    const float* g2 = (const float*)d_in[8];
    float* out = (float*)d_out;

    const size_t MB = 1ull << 20;
    char* ws = (char*)d_ws;
    bf16* Wqb = (bf16*)(ws + 0 * MB);
    bf16* Wkb = (bf16*)(ws + 2 * MB);
    bf16* Wvb = (bf16*)(ws + 4 * MB);
    bf16* Wob = (bf16*)(ws + 6 * MB);
    bf16* W1b = (bf16*)(ws + 8 * MB);
    bf16* W2b = (bf16*)(ws + 16 * MB);
    bf16* hb  = (bf16*)(ws + 24 * MB);   // reused later as h2
    bf16* qb  = (bf16*)(ws + 32 * MB);
    bf16* kbp = (bf16*)(ws + 40 * MB);
    bf16* vb  = (bf16*)(ws + 48 * MB);
    bf16* vt  = (bf16*)(ws + 56 * MB);
    bf16* ab  = (bf16*)(ws + 64 * MB);
    float* y  = (float*)(ws + 72 * MB);  // 16 MB
    bf16* ff1 = qb;                       // 32 MB region qb..vt reused

    auto cvt = [&](const float* src, bf16* dst, int n) {
        int n4 = n / 4;
        cvt_f32_bf16<<<(n4 + 255) / 256, 256, 0, stream>>>(src, dst, n4);
    };
    cvt(Wq, Wqb, DMODEL * DMODEL);
    cvt(Wk, Wkb, DMODEL * DMODEL);
    cvt(Wv, Wvb, DMODEL * DMODEL);
    cvt(Wo, Wob, DMODEL * DMODEL);
    cvt(W1, W1b, DFF * DMODEL);
    cvt(W2, W2b, DMODEL * DFF);

    // h = rmsnorm(x, g1)
    rmsnorm_bf16<<<NTOK, 256, 0, stream>>>(x, g1, hb);

    // q/k/v = h @ W^T
    dim3 gqkv(DMODEL / 128, NTOK / 128);
    gemm_wmma_bf16<0><<<gqkv, 256, 0, stream>>>(hb, Wqb, qb, nullptr, nullptr, NTOK, DMODEL, DMODEL);
    gemm_wmma_bf16<0><<<gqkv, 256, 0, stream>>>(hb, Wkb, kbp, nullptr, nullptr, NTOK, DMODEL, DMODEL);
    gemm_wmma_bf16<0><<<gqkv, 256, 0, stream>>>(hb, Wvb, vb, nullptr, nullptr, NTOK, DMODEL, DMODEL);

    // V transpose for contiguous B-operand loads in PV matmul
    transpose_v<<<(NTOK * DMODEL) / 256, 256, 0, stream>>>(vb, vt);

    // attention (4096 wave-tiles, 8 waves/block)
    attn_wmma<<<512, 256, 0, stream>>>(qb, kbp, vt, ab);

    // y = x + attn @ Wo^T
    gemm_wmma_bf16<2><<<gqkv, 256, 0, stream>>>(ab, Wob, nullptr, y, x, NTOK, DMODEL, DMODEL);

    // h2 = rmsnorm(y, g2)
    rmsnorm_bf16<<<NTOK, 256, 0, stream>>>(y, g2, hb);

    // ff1 = gelu(h2 @ W1^T)
    dim3 gff1(DFF / 128, NTOK / 128);
    gemm_wmma_bf16<1><<<gff1, 256, 0, stream>>>(hb, W1b, ff1, nullptr, nullptr, NTOK, DFF, DMODEL);

    // out = y + ff1 @ W2^T
    gemm_wmma_bf16<2><<<gqkv, 256, 0, stream>>>(ff1, W2b, nullptr, out, y, NTOK, DMODEL, DFF);
}
